// TopKSAE_34016140985002
// MI455X (gfx1250) — matrix operand order repaired and checked by reading, not compiled, not run on hardware
//
#include <hip/hip_runtime.h>

#define NROWS 8192
#define DIN   1024
#define DHID  16384
#define KTOP  32

typedef __attribute__((ext_vector_type(16))) __bf16        v16bf;
typedef __attribute__((ext_vector_type(8)))  float         v8f;
typedef __attribute__((ext_vector_type(8)))  unsigned int  v8u;

union FragAB {
    struct { uint4 q0, q1; } q;
    v8u   u;
    v16bf b;
};

__device__ __forceinline__ unsigned short f32_to_bf16_rne(float f) {
    unsigned int u = __float_as_uint(f);
    unsigned int r = u + 0x7FFFu + ((u >> 16) & 1u);
    return (unsigned short)(r >> 16);
}

__device__ __forceinline__ void split_bf16(float f, unsigned short& h, unsigned short& l) {
    h = f32_to_bf16_rne(f);
    float hf = __uint_as_float(((unsigned int)h) << 16);
    l = f32_to_bf16_rne(f - hf);
}

// ---------------------------------------------------------------------------
// Kernel A: z = relu(x @ W_enc^T) with bf16-split-3 WMMA (hi*hi + hi*lo + lo*hi)
// Block: 256 threads (8 waves). Tile: 128 rows (x) x 128 cols (hidden), BK=32.
// ---------------------------------------------------------------------------
__global__ __launch_bounds__(256)
void sae_encoder_wmma(const float* __restrict__ x,
                      const float* __restrict__ W_enc,
                      float* __restrict__ z) {
    __shared__ __align__(32) unsigned short As_hi[128 * 32];
    __shared__ __align__(32) unsigned short As_lo[128 * 32];
    __shared__ __align__(32) unsigned short Bs_hi[128 * 32];
    __shared__ __align__(32) unsigned short Bs_lo[128 * 32];

    const int tid  = threadIdx.x;
    const int lane = tid & 31;
    const int wave = tid >> 5;
    const int wr   = wave & 3;    // 4 row-groups of 32 rows
    const int wc   = wave >> 2;   // 2 col-groups of 64 cols
    const int rowBase = blockIdx.y * 128;
    const int colBase = blockIdx.x * 128;

    v8f acc[2][4];
#pragma unroll
    for (int mi = 0; mi < 2; mi++)
#pragma unroll
        for (int ni = 0; ni < 4; ni++)
            acc[mi][ni] = (v8f){0.f,0.f,0.f,0.f,0.f,0.f,0.f,0.f};

    const int r8 = tid >> 3;        // 0..31: row within 32-row slab
    const int c4 = (tid & 7) * 4;   // 0,4,..,28: K column (float4 granule)

    for (int kk = 0; kk < DIN; kk += 32) {
        // speculative prefetch of next K-tile (global_prefetch_b8)
        __builtin_prefetch(x     + (size_t)(rowBase + r8) * DIN + kk + 32 + c4, 0, 1);
        __builtin_prefetch(W_enc + (size_t)(colBase + r8) * DIN + kk + 32 + c4, 0, 1);

        // Stage A (x) and B (W_enc) K-slabs into LDS as split bf16 hi/lo
#pragma unroll
        for (int i = 0; i < 4; i++) {
            const int row = r8 + i * 32;
            const float4 av = *(const float4*)(x     + (size_t)(rowBase + row) * DIN + kk + c4);
            const float4 bv = *(const float4*)(W_enc + (size_t)(colBase + row) * DIN + kk + c4);
            unsigned short h0,h1,h2,h3,l0,l1,l2,l3;

            split_bf16(av.x,h0,l0); split_bf16(av.y,h1,l1);
            split_bf16(av.z,h2,l2); split_bf16(av.w,h3,l3);
            *(uint2*)&As_hi[row*32 + c4] =
                make_uint2((unsigned)h0 | ((unsigned)h1 << 16), (unsigned)h2 | ((unsigned)h3 << 16));
            *(uint2*)&As_lo[row*32 + c4] =
                make_uint2((unsigned)l0 | ((unsigned)l1 << 16), (unsigned)l2 | ((unsigned)l3 << 16));

            split_bf16(bv.x,h0,l0); split_bf16(bv.y,h1,l1);
            split_bf16(bv.z,h2,l2); split_bf16(bv.w,h3,l3);
            *(uint2*)&Bs_hi[row*32 + c4] =
                make_uint2((unsigned)h0 | ((unsigned)h1 << 16), (unsigned)h2 | ((unsigned)h3 << 16));
            *(uint2*)&Bs_lo[row*32 + c4] =
                make_uint2((unsigned)l0 | ((unsigned)l1 << 16), (unsigned)l2 | ((unsigned)l3 << 16));
        }
        __syncthreads();

        // Per-lane fragment loads; A and B both use contiguous k order (k = 16*(lane/16) + s).
        FragAB a_hi[2], a_lo[2], b_hi[4], b_lo[4];
#pragma unroll
        for (int mi = 0; mi < 2; mi++) {
            const int mrow = wr * 32 + mi * 16 + (lane & 15);
            const int kb   = (lane >> 4) * 16;    // lanes 0-15: K 0..15; 16-31: K 16..31
            const int base = mrow * 32 + kb;
            a_hi[mi].q.q0 = *(const uint4*)&As_hi[base];
            a_hi[mi].q.q1 = *(const uint4*)&As_hi[base + 8];
            a_lo[mi].q.q0 = *(const uint4*)&As_lo[base];
            a_lo[mi].q.q1 = *(const uint4*)&As_lo[base + 8];
        }
#pragma unroll
        for (int ni = 0; ni < 4; ni++) {
            const int ncol = wc * 64 + ni * 16 + (lane & 15);
            const int kb   = (lane >> 4) * 16;    // lanes 0-15: K 0..15; 16-31: K 16..31
            const int base = ncol * 32 + kb;
            b_hi[ni].u = *(const v8u*)&Bs_hi[base];
            b_lo[ni].u = *(const v8u*)&Bs_lo[base];
        }

#pragma unroll
        for (int mi = 0; mi < 2; mi++)
#pragma unroll
            for (int ni = 0; ni < 4; ni++) {
                acc[mi][ni] = __builtin_amdgcn_wmma_f32_16x16x32_bf16(
                    false, a_hi[mi].b, false, b_hi[ni].b, (short)0, acc[mi][ni], false, false);
                acc[mi][ni] = __builtin_amdgcn_wmma_f32_16x16x32_bf16(
                    false, a_hi[mi].b, false, b_lo[ni].b, (short)0, acc[mi][ni], false, false);
                acc[mi][ni] = __builtin_amdgcn_wmma_f32_16x16x32_bf16(
                    false, a_lo[mi].b, false, b_hi[ni].b, (short)0, acc[mi][ni], false, false);
            }
        __syncthreads();
    }

    // Epilogue: ReLU + store z. C/D layout: lanes 0-15 -> M=j, lanes 16-31 -> M=8+j.
#pragma unroll
    for (int mi = 0; mi < 2; mi++)
#pragma unroll
        for (int ni = 0; ni < 4; ni++) {
            const int n  = colBase + wc * 64 + ni * 16 + (lane & 15);
            const int m0 = rowBase + wr * 32 + mi * 16 + ((lane >> 4) << 3);
#pragma unroll
            for (int j = 0; j < 8; j++)
                z[(size_t)(m0 + j) * DHID + n] = fmaxf(acc[mi][ni][j], 0.0f);
        }
}

// ---------------------------------------------------------------------------
// Kernel B: per-row top-32 selection, z_sparse scatter, decoder gather, active.
// One block (256 threads) per row; row lives in 64 registers per thread.
// ---------------------------------------------------------------------------
__global__ __launch_bounds__(256)
void sae_topk_decode(const float* __restrict__ z,
                     const float* __restrict__ W_dec,
                     float* __restrict__ z_sparse,
                     float* __restrict__ x_hat,
                     float* __restrict__ active) {
    __shared__ float smax[256];
    __shared__ int   sidx[256];
    __shared__ float topv[KTOP];
    __shared__ int   topi[KTOP];

    const int tid = threadIdx.x;
    const int row = blockIdx.x;
    const float* zr = z + (size_t)row * DHID;

    float rv[64];
#pragma unroll
    for (int c = 0; c < 16; c++) {
        const float4 v = *(const float4*)(zr + c * 1024 + tid * 4);
        rv[c*4+0] = v.x; rv[c*4+1] = v.y; rv[c*4+2] = v.z; rv[c*4+3] = v.w;
    }

    float* zsr = z_sparse + (size_t)row * DHID;

    for (int r = 0; r < KTOP; r++) {
        float m = -1e30f; int mq = 0;
#pragma unroll
        for (int q = 0; q < 64; q++)
            if (rv[q] > m) { m = rv[q]; mq = q; }
        smax[tid] = m;
        sidx[tid] = (mq >> 2) * 1024 + tid * 4 + (mq & 3);
        __syncthreads();
        for (int s = 128; s > 0; s >>= 1) {
            if (tid < s && smax[tid + s] > smax[tid]) {
                smax[tid] = smax[tid + s];
                sidx[tid] = sidx[tid + s];
            }
            __syncthreads();
        }
        const int   wj = sidx[0];
        const float wv = smax[0];
        if (tid == 0) { topv[r] = wv; topi[r] = wj; }
        if (tid == ((wj & 1023) >> 2)) {          // owning thread removes + scatters
            const int local = ((wj >> 10) << 2) | (wj & 3);
#pragma unroll
            for (int q = 0; q < 64; q++)
                if (q == local) rv[q] = -1e30f;
            zsr[wj] = wv;
        }
        __syncthreads();
    }

    // Zero the unselected positions (selected ones were already scattered).
#pragma unroll
    for (int c = 0; c < 16; c++)
#pragma unroll
        for (int e = 0; e < 4; e++)
            if (rv[c*4+e] != -1e30f)
                zsr[c * 1024 + tid * 4 + e] = 0.0f;

    // Decoder: x_hat[row,:] = sum_k topv[k] * W_dec[:, topi[k]]  (W_dec L2-resident)
    float a0 = 0.f, a1 = 0.f, a2 = 0.f, a3 = 0.f;
    const int i0 = tid * 4;
#pragma unroll 8
    for (int t = 0; t < KTOP; t++) {
        const float w = topv[t];
        const float* wd = W_dec + (size_t)topi[t];
        a0 += w * wd[(size_t)(i0 + 0) * DHID];
        a1 += w * wd[(size_t)(i0 + 1) * DHID];
        a2 += w * wd[(size_t)(i0 + 2) * DHID];
        a3 += w * wd[(size_t)(i0 + 3) * DHID];
    }
    *(float4*)(x_hat + (size_t)row * DIN + i0) = make_float4(a0, a1, a2, a3);

    if (tid == 0) {
        int cnt = 0;
#pragma unroll
        for (int t = 0; t < KTOP; t++) cnt += (topv[t] > 0.0f) ? 1 : 0;
        atomicAdd(active, (float)cnt * (1.0f / (float)NROWS));
    }
}

__global__ void sae_init_active(float* active) { *active = 0.0f; }

// ---------------------------------------------------------------------------
extern "C" void kernel_launch(void* const* d_in, const int* in_sizes, int n_in,
                              void* d_out, int out_size, void* d_ws, size_t ws_size,
                              hipStream_t stream) {
    const float* x     = (const float*)d_in[0];
    const float* W_enc = (const float*)d_in[1];
    const float* W_dec = (const float*)d_in[2];

    float* out      = (float*)d_out;
    float* x_hat    = out;                                      // [NROWS, DIN]
    float* z_sparse = x_hat + (size_t)NROWS * DIN;              // [NROWS, DHID]
    float* z        = z_sparse + (size_t)NROWS * DHID;          // [NROWS, DHID]
    float* active   = z + (size_t)NROWS * DHID;                 // scalar

    dim3 gridA(DHID / 128, NROWS / 128);
    sae_encoder_wmma<<<gridA, 256, 0, stream>>>(x, W_enc, z);
    sae_init_active<<<1, 1, 0, stream>>>(active);
    sae_topk_decode<<<NROWS, 256, 0, stream>>>(z, W_dec, z_sparse, x_hat, active);
}
